// GRU_87540023427964
// MI455X (gfx1250) — matrix operand order repaired and checked by reading, not compiled, not run on hardware
//
#include <hip/hip_runtime.h>
#include <hip/hip_bf16.h>

// ---------------------------------------------------------------------------
// GRU (B=64, T=1024, D=512, H=1024, O=256) for gfx1250 (MI455X).
//
//   1) x -> bf16
//   2) pack weights into V_WMMA_F32_16X16X32_BF16 B-fragment layout
//   3) time-parallel x-part preactivations (whole-chip WMMA GEMM)
//   4) persistent scan: 4 workgroups, one 16-row batch slice each; h / r*h / u
//      in LDS; 2 s_barriers per timestep; 2-stage software-pipelined K-loops.
//
// Round-5 changes: round 4 spilled the WMMA accumulators inside the gate
// loop because __launch_bounds__(1024,1) (8 waves/SIMD) caps per-wave VGPRs
// at ~128, right at the J=4 pipeline's ~130 live registers. The scan now
// uses 512 threads (16 waves = 4 waves/SIMD -> ~256 VGPR cap):
//   Phase A = two sequential J=4 MACs per wave (r-half then u-half, no
//   branch), Phase B = one J=4 MAC per wave (halves redundant LDS A reads),
//   projection = exactly one tile per wave. Same total WMMA work, no spills.
// ---------------------------------------------------------------------------

#define Bsz 64
#define Tsz 1024
#define Dsz 512
#define Hsz 1024
#define Osz 256

typedef __attribute__((ext_vector_type(16))) __bf16    v16bf;
typedef __attribute__((ext_vector_type(8)))  float     v8f;
typedef __attribute__((ext_vector_type(4)))  unsigned  v4u;

union Frag { v16bf bf; v4u u[2]; };

#define WMMA_BF16(a, b, c) \
  __builtin_amdgcn_wmma_f32_16x16x32_bf16(false, (a), false, (b), (short)0, (c), false, false)

__device__ __forceinline__ float sigmf(float x) {
  return 1.0f / (1.0f + __expf(-x));
}
__device__ __forceinline__ float tanh_fast(float x) {
  float t = __expf(-2.0f * fabsf(x));
  float r = (1.0f - t) / (1.0f + t);
  return copysignf(r, x);
}

// A fragment (16x32 bf16): laneBase = src + (lane&15)*stride + (lane>>4)*8,
// chunk0 at +kElem, chunk1 at +kElem+16 elements.
__device__ __forceinline__ Frag loadA_off(const __bf16* laneBase, int kElem) {
  Frag f;
  f.u[0] = *(const v4u*)(laneBase + kElem);
  f.u[1] = *(const v4u*)(laneBase + kElem + 16);
  return f;
}

// B fragment from packed layout: laneBase = P + lane*32 (+ nt0*1024) bytes.
__device__ __forceinline__ Frag loadB_off(const char* laneBase, int byteOff) {
  Frag f;
  f.u[0] = *(const v4u*)(laneBase + byteOff);
  f.u[1] = *(const v4u*)(laneBase + byteOff + 16);
  return f;
}

// ---------------------------------------------------------------------------
// 2-stage software-pipelined WMMA MAC over KT K-tiles of J N-tiles each.
// bBase points at the kt=0 origin of this GEMM's packed weights; KSTRIDE is
// the byte stride between consecutive kt fragment groups (NT16 * 1024).
// Live set: acc[J] + 2*(A frag + J B frags).
// ---------------------------------------------------------------------------
template <int J, int KT, int KSTRIDE>
__device__ __forceinline__ void wmma_mac(v8f* acc, const __bf16* aBase,
                                         const char* bBase) {
  Frag a0, a1, b0[J], b1[J];
  a0 = loadA_off(aBase, 0);
#pragma unroll
  for (int j = 0; j < J; ++j) b0[j] = loadB_off(bBase, j * 1024);
#pragma unroll 1
  for (int kt = 0; kt < KT - 2; kt += 2) {
    a1 = loadA_off(aBase, (kt + 1) * 32);
#pragma unroll
    for (int j = 0; j < J; ++j)
      b1[j] = loadB_off(bBase, (kt + 1) * KSTRIDE + j * 1024);
#pragma unroll
    for (int j = 0; j < J; ++j) acc[j] = WMMA_BF16(a0.bf, b0[j].bf, acc[j]);
    a0 = loadA_off(aBase, (kt + 2) * 32);
#pragma unroll
    for (int j = 0; j < J; ++j)
      b0[j] = loadB_off(bBase, (kt + 2) * KSTRIDE + j * 1024);
#pragma unroll
    for (int j = 0; j < J; ++j) acc[j] = WMMA_BF16(a1.bf, b1[j].bf, acc[j]);
  }
  // tail: kt = KT-2 (already in a0/b0), KT-1
  a1 = loadA_off(aBase, (KT - 1) * 32);
#pragma unroll
  for (int j = 0; j < J; ++j)
    b1[j] = loadB_off(bBase, (KT - 1) * KSTRIDE + j * 1024);
#pragma unroll
  for (int j = 0; j < J; ++j) acc[j] = WMMA_BF16(a0.bf, b0[j].bf, acc[j]);
#pragma unroll
  for (int j = 0; j < J; ++j) acc[j] = WMMA_BF16(a1.bf, b1[j].bf, acc[j]);
}

// ---------------------------------------------------------------------------
__global__ void cvt_bf16_kernel(const float* __restrict__ src,
                                __bf16* __restrict__ dst, long n) {
  long i = (long)blockIdx.x * blockDim.x + threadIdx.x;
  if (i < n) dst[i] = (__bf16)src[i];
}

// Pack fp32 row-major W[K,N] into bf16 WMMA B-fragment tiles.
__global__ void pack_b_kernel(const float* __restrict__ W,
                              unsigned* __restrict__ P, int K, int N) {
  long idx = (long)blockIdx.x * blockDim.x + threadIdx.x;
  long total = ((long)K * N) >> 1;
  if (idx >= total) return;
  int v    = idx & 7;
  int lane = (idx >> 3) & 31;
  long tile = idx >> 8;
  int NT16 = N >> 4;
  int nt = (int)(tile % NT16);
  int kt = (int)(tile / NT16);
  int k = kt * 32 + ((lane >> 4) << 3) + ((v >> 2) << 4) + 2 * (v & 3);
  int n = nt * 16 + (lane & 15);
  union { unsigned u; __bf16 b[2]; } d;
  d.b[0] = (__bf16)W[(size_t)k * N + n];
  d.b[1] = (__bf16)W[(size_t)(k + 1) * N + n];
  P[idx] = d.u;
}

// Time-parallel precompute: C[Mrows,N] (bf16) = A[Mrows,512] @ Bpacked.
template <int N>
__global__ void __launch_bounds__(256, 1)
gemm_pre_kernel(const __bf16* __restrict__ A, const unsigned* __restrict__ Bp,
                __bf16* __restrict__ C, int Mrows) {
  constexpr int NT16 = N >> 4;
  int flat = blockIdx.x * blockDim.x + threadIdx.x;
  int wave = flat >> 5;
  int lane = flat & 31;
  int ntStrips = N >> 6;
  int mt  = wave / ntStrips;
  int nt0 = (wave % ntStrips) << 2;
  if (mt >= (Mrows >> 4)) return;
  int row0 = mt << 4;

  const __bf16* aBase = A + (size_t)(row0 + (lane & 15)) * Dsz + ((lane >> 4) << 3);
  const char*   bBase = (const char*)Bp + (size_t)lane * 32 + (size_t)nt0 * 1024;

  v8f acc[4] = {};
  wmma_mac<4, 16, NT16 * 1024>(acc, aBase, bBase);   // K = 512 (x-part rows)

  const int hi = lane >> 4, nl = lane & 15;
#pragma unroll
  for (int j = 0; j < 4; ++j) {
    int n = (nt0 + j) * 16 + nl;
#pragma unroll
    for (int i = 0; i < 8; ++i) {
      int m = row0 + i + (hi << 3);
      C[(size_t)m * N + n] = (__bf16)acc[j][i];
    }
  }
}

// ---------------------------------------------------------------------------
// Persistent GRU scan. grid=4 blocks (one 16-row batch slice each),
// 512 threads = 16 waves on one WGP (4 waves/SIMD -> ample VGPR budget).
// LDS: h(32K) + r*h(32K) + u(64K).
// ---------------------------------------------------------------------------
__global__ void __launch_bounds__(512, 1)
gru_scan_kernel(const __bf16* __restrict__ xg, const __bf16* __restrict__ xc,
                const unsigned* __restrict__ Wg, const unsigned* __restrict__ Wc,
                const unsigned* __restrict__ Wp,
                const float* __restrict__ gbias, const float* __restrict__ cbias,
                const float* __restrict__ pbias, float* __restrict__ out) {
  extern __shared__ char smem[];
  __bf16* hs  = (__bf16*)smem;            // [16][1024] h_t         (32 KB)
  __bf16* rhs = hs + 16 * Hsz;            // [16][1024] r*h         (32 KB)
  float*  us  = (float*)(rhs + 16 * Hsz); // [16][1024] update gate (64 KB)

  const int g    = blockIdx.x;
  const int tid  = threadIdx.x;
  const int wave = tid >> 5;              // 0..15
  const int lane = tid & 31;
  const int hi   = lane >> 4;
  const int nl   = lane & 15;
  const int b0   = g * 16;

  // Per-lane fragment base pointers (constant across the whole scan).
  const __bf16* aH  = hs  + (size_t)nl * Hsz + (hi << 3);
  const __bf16* aRH = rhs + (size_t)nl * Hsz + (hi << 3);
  const char*   bWg = (const char*)Wg + (size_t)lane * 32;   // NT16 = 128
  const char*   bWc = (const char*)Wc + (size_t)lane * 32;   // NT16 = 64
  const char*   bWp = (const char*)Wp + (size_t)lane * 32;   // NT16 = 16

  for (int i = tid; i < 16 * Hsz; i += 512) hs[i] = (__bf16)0.0f;
  __syncthreads();

  for (int t = 0; t < Tsz; ++t) {
    // ---------------- Phase A: gates (+ projection of h_{t-1}) ------------
    {
      // --- reset-gate half: tiles nt = wave*4 .. wave*4+3 (0..63) ---------
      {
        const int nt0 = wave << 2;
        v8f acc[4] = {};
        wmma_mac<4, 32, 128 * 1024>(acc, aH,
                                    bWg + (size_t)nt0 * 1024 + 16 * (128 * 1024));
#pragma unroll
        for (int j = 0; j < 4; ++j) {
          int n = (nt0 + j) * 16 + nl;
          float bb = gbias[n];
#pragma unroll
          for (int i = 0; i < 8; ++i) {
            int m = i + (hi << 3);
            float pre = acc[j][i] + bb +
                (float)xg[((size_t)(b0 + m) * Tsz + t) * (2 * Hsz) + n];
            float hv = (float)hs[m * Hsz + n];
            rhs[m * Hsz + n] = (__bf16)(sigmf(pre) * hv);
          }
        }
      }
      // --- update-gate half: tiles nt = 64 + wave*4 .. +3 (64..127) -------
      {
        const int nt0 = 64 + (wave << 2);
        v8f acc[4] = {};
        wmma_mac<4, 32, 128 * 1024>(acc, aH,
                                    bWg + (size_t)nt0 * 1024 + 16 * (128 * 1024));
#pragma unroll
        for (int j = 0; j < 4; ++j) {
          int n  = (nt0 + j) * 16 + nl;
          int n2 = n - Hsz;
          float bb = gbias[n];
#pragma unroll
          for (int i = 0; i < 8; ++i) {
            int m = i + (hi << 3);
            float pre = acc[j][i] + bb +
                (float)xg[((size_t)(b0 + m) * Tsz + t) * (2 * Hsz) + n];
            us[m * Hsz + n2] = sigmf(pre);
          }
        }
      }
      // --- fused projection of h_{t-1} -> out[:, t-1, :] (1 tile/wave) ----
      if (t > 0) {
        const int nt = wave;
        const int n  = nt * 16 + nl;
        v8f pa[1] = {};
        wmma_mac<1, 32, 16 * 1024>(pa, aH, bWp + (size_t)nt * 1024);
        float pb = pbias[n];
#pragma unroll
        for (int i = 0; i < 8; ++i) {
          int m = i + (hi << 3);
          out[((size_t)(b0 + m) * Tsz + (t - 1)) * Osz + n] = sigmf(pa[0][i] + pb);
        }
      }
      // prefetch next timestep's gate preactivations (global_prefetch_b8)
      if (t + 1 < Tsz) {
#pragma unroll
        for (int i = 0; i < 8; ++i) {
          int m = i + (hi << 3);
          const __bf16* rowp =
              &xg[((size_t)(b0 + m) * Tsz + (t + 1)) * (2 * Hsz)];
          __builtin_prefetch(rowp + (wave << 6) + nl, 0, 1);
          __builtin_prefetch(rowp + Hsz + (wave << 6) + nl, 0, 1);
        }
      }
    }
    __syncthreads();
    // ---------------- Phase B: candidate + h update -----------------------
    {
      const int nt0 = wave << 2;                    // 4 of 64 cand tiles
      v8f acc[4] = {};
      wmma_mac<4, 32, 64 * 1024>(acc, aRH,
                                 bWc + (size_t)nt0 * 1024 + 16 * (64 * 1024));
#pragma unroll
      for (int j = 0; j < 4; ++j) {
        int n = (nt0 + j) * 16 + nl;
        float bb = cbias[n];
#pragma unroll
        for (int i = 0; i < 8; ++i) {
          int m = i + (hi << 3);
          float pre = acc[j][i] + bb +
              (float)xc[((size_t)(b0 + m) * Tsz + t) * Hsz + n];
          float c  = tanh_fast(pre);
          float uu = us[m * Hsz + n];
          float hv = (float)hs[m * Hsz + n];
          hs[m * Hsz + n] = (__bf16)(uu * hv + (1.0f - uu) * c);
        }
      }
      if (t + 1 < Tsz) {
#pragma unroll
        for (int i = 0; i < 8; ++i) {
          int m = i + (hi << 3);
          __builtin_prefetch(
              &xc[((size_t)(b0 + m) * Tsz + (t + 1)) * Hsz + (wave << 6) + nl], 0, 1);
        }
      }
    }
    __syncthreads();
  }
  // final projection for t = T-1 (1 tile/wave)
  {
    const int nt = wave;
    const int n  = nt * 16 + nl;
    v8f pa[1] = {};
    wmma_mac<1, 32, 16 * 1024>(pa, aH, bWp + (size_t)nt * 1024);
    float pb = pbias[n];
#pragma unroll
    for (int i = 0; i < 8; ++i) {
      int m = i + (hi << 3);
      out[((size_t)(b0 + m) * Tsz + (Tsz - 1)) * Osz + n] = sigmf(pa[0][i] + pb);
    }
  }
}

// ---------------------------------------------------------------------------
extern "C" void kernel_launch(void* const* d_in, const int* in_sizes, int n_in,
                              void* d_out, int out_size, void* d_ws, size_t ws_size,
                              hipStream_t stream) {
  (void)in_sizes; (void)n_in; (void)out_size; (void)ws_size;
  const float* x  = (const float*)d_in[0];
  const float* gk = (const float*)d_in[1];
  const float* gb = (const float*)d_in[2];
  const float* ck = (const float*)d_in[3];
  const float* cb = (const float*)d_in[4];
  const float* pw = (const float*)d_in[5];
  const float* pb = (const float*)d_in[6];
  float* out = (float*)d_out;

  char* ws = (char*)d_ws;
  size_t off = 0;
  auto take = [&](size_t bytes) -> char* {
    char* p = ws + off;
    off = (off + bytes + 255) & ~(size_t)255;
    return p;
  };
  __bf16*   xbf = (__bf16*)take((size_t)Bsz * Tsz * Dsz * 2);           //  67 MB
  unsigned* Wg  = (unsigned*)take((size_t)(Dsz + Hsz) * 2 * Hsz * 2);   // 6.3 MB
  unsigned* Wc  = (unsigned*)take((size_t)(Dsz + Hsz) * Hsz * 2);       // 3.1 MB
  unsigned* Wp  = (unsigned*)take((size_t)Hsz * Osz * 2);               // 0.5 MB
  __bf16*   xg  = (__bf16*)take((size_t)Bsz * Tsz * 2 * Hsz * 2);       // 268 MB
  __bf16*   xc  = (__bf16*)take((size_t)Bsz * Tsz * Hsz * 2);           // 134 MB

  // 1) x -> bf16
  long nx = (long)Bsz * Tsz * Dsz;
  cvt_bf16_kernel<<<(nx + 255) / 256, 256, 0, stream>>>(x, xbf, nx);

  // 2) pack weights into WMMA B-fragment layout
  long ng = (long)(Dsz + Hsz) * (2 * Hsz) / 2;
  long nc = (long)(Dsz + Hsz) * Hsz / 2;
  long np = (long)Hsz * Osz / 2;
  pack_b_kernel<<<(ng + 255) / 256, 256, 0, stream>>>(gk, Wg, Dsz + Hsz, 2 * Hsz);
  pack_b_kernel<<<(nc + 255) / 256, 256, 0, stream>>>(ck, Wc, Dsz + Hsz, Hsz);
  pack_b_kernel<<<(np + 255) / 256, 256, 0, stream>>>(pw, Wp, Hsz, Osz);

  // 3) time-parallel x-part preactivations: M = B*T = 65536 rows
  {
    int Mrows = Bsz * Tsz;
    int wavesG = (Mrows >> 4) * ((2 * Hsz) >> 6);   // 4096 * 32
    int wavesC = (Mrows >> 4) * (Hsz >> 6);         // 4096 * 16
    gemm_pre_kernel<2 * Hsz><<<wavesG * 32 / 256, 256, 0, stream>>>(xbf, Wg, xg, Mrows);
    gemm_pre_kernel<Hsz><<<wavesC * 32 / 256, 256, 0, stream>>>(xbf, Wc, xc, Mrows);
  }

  // 4) persistent scan: 4 batch slices, 128 KB dynamic LDS each
  (void)hipFuncSetAttribute((const void*)gru_scan_kernel,
                            hipFuncAttributeMaxDynamicSharedMemorySize, 131072);
  gru_scan_kernel<<<4, 512, 131072, stream>>>(xg, xc, Wg, Wc, Wp, gb, cb, pb, out);
}
